// Mamba_4303557231107
// MI455X (gfx1250) — compile-verified, matrix operand order checked by reading, not run on hardware
//
#include <hip/hip_runtime.h>
#include <hip/hip_bf16.h>

#define D_MODEL 768
#define N_LAYER 4
#define D_STATE 16
#define D_CONV  4
#define DT_RANK 48
#define D_INNER 1536
#define B_SZ    2
#define L_SEQ   1024
#define N_MELS  80
#define XDBL_N  (DT_RANK + 2 * D_STATE)   /* 80 */
#define ROWS    (B_SZ * L_SEQ)            /* 2048 */

typedef __bf16 v16bf __attribute__((ext_vector_type(16)));
typedef __bf16 v8bf  __attribute__((ext_vector_type(8)));
typedef float  v8f   __attribute__((ext_vector_type(8)));

__device__ __forceinline__ float softplus_f(float x) {
  return (x > 20.f) ? x : log1pf(expf(x));
}
__device__ __forceinline__ float silu_f(float x) {
  return x / (1.f + expf(-x));
}

enum { EPI_NONE = 0, EPI_BIAS_SOFTPLUS = 1, EPI_RESID = 2 };

// ---------------------------------------------------------------------------
// Weight pre-pack: f32 K x N (row-major) -> fragment-major bf16.
// Fragment (kb, ng) = 32x16 bf16 block in CDNA5 WMMA B-operand layout,
// 512 bf16 contiguous; each lane's 16 elements contiguous (32B).
// K is zero-padded to KB*32 so the GEMM needs no tail handling.
// One thread per (fragment, lane): 16 strided reads, one 32B store.
// ---------------------------------------------------------------------------
__global__ __launch_bounds__(256) void pack_b_kernel(
    const float* __restrict__ Bw, __bf16* __restrict__ pk, int K, int N,
    int KB) {
  int tid = blockIdx.x * blockDim.x + threadIdx.x;
  int nG = N >> 4;
  int total = KB * nG * 32;
  if (tid >= total) return;
  int lane = tid & 31;
  int frag = tid >> 5;
  int kb = frag / nG, ng = frag - kb * nG;
  int col  = ng * 16 + (lane & 15);
  int lohi = (lane >= 16) ? 8 : 0;
  int k0   = kb * 32;
  v16bf f;
#pragma unroll
  for (int e = 0; e < 16; ++e) {
    int klocal = (e < 8) ? (lohi + e) : (16 + lohi + (e - 8));
    int k  = k0 + klocal;
    int kc = (k < K) ? k : (K - 1);           // clamped -> always-valid load
    float v = Bw[(size_t)kc * N + col];
    f[e] = (__bf16)((k < K) ? v : 0.f);       // select, no branch
  }
  *(v16bf*)(pk + (size_t)tid * 16) = f;
}

// ---------------------------------------------------------------------------
// A fragment: bf16 row-major A, per lane two contiguous 8-element runs
// (16B each) at k0+lohi and k0+16+lohi.
// ---------------------------------------------------------------------------
__device__ __forceinline__ v16bf load_a16(const __bf16* __restrict__ p) {
  v8bf lo = *(const v8bf*)(p);
  v8bf hi = *(const v8bf*)(p + 16);
  return __builtin_shufflevector(lo, hi, 0, 1, 2, 3, 4, 5, 6, 7, 8, 9, 10, 11,
                                 12, 13, 14, 15);
}

// ---------------------------------------------------------------------------
// WMMA GEMM: C[M,N](f32) = A[M,*](bf16, lda) * Bpk (fragment-major bf16).
// One wave computes a 32 x (16*NT) strip (2 M-tiles reuse each B fragment).
// KB = number of 32-wide K steps (B is zero-padded). Per k-step:
// 4 b128 A loads + 2*NT b128 B loads + 2*NT WMMAs, zero conversions.
// EXEC stays all-ones for every live wave (launcher guarantees divisibility).
// ---------------------------------------------------------------------------
template <int EPI, int NT, int KB>
__global__ __launch_bounds__(256) void gemm_wmma_pk(
    const __bf16* __restrict__ A, int lda,
    const __bf16* __restrict__ Bpk, int nG /* N/16 */,
    float* __restrict__ C, int ldc,
    int M, int N,
    const float* __restrict__ bias,
    const float* __restrict__ resid, int ldres) {
  const int lane = threadIdx.x & 31;
  const int wave = blockIdx.x * 8 + (threadIdx.x >> 5);
  const int tilesNG = N / (16 * NT);
  const int totalWaves = (M >> 5) * tilesNG;
  if (wave >= totalWaves) return;  // uniform per wave (never trips)

  const int tm   = wave / tilesNG;
  const int tng  = wave % tilesNG;
  const bool hi  = lane >= 16;
  const int lrow = lane & 15;
  const int lohi = hi ? 8 : 0;

  v8f zero = {};
  v8f acc0[NT], acc1[NT];
#pragma unroll
  for (int j = 0; j < NT; ++j) { acc0[j] = zero; acc1[j] = zero; }

  const __bf16* __restrict__ a0p = A + (size_t)(tm * 32 + lrow) * lda + lohi;
  const __bf16* __restrict__ a1p = a0p + (size_t)16 * lda;
  // Packed B: fragment (kb, g) at (kb*nG + g)*512, lane offset lane*16.
  const __bf16* __restrict__ bb =
      Bpk + (size_t)(tng * NT) * 512 + (size_t)lane * 16;
  const size_t kstep = (size_t)nG * 512;

  for (int kb = 0; kb < KB; ++kb) {
    v16bf fa0 = load_a16(a0p + kb * 32);
    v16bf fa1 = load_a16(a1p + kb * 32);
    const __bf16* bk = bb + (size_t)kb * kstep;
#pragma unroll
    for (int j = 0; j < NT; ++j) {
      v16bf fb = *(const v16bf*)(bk + j * 512);
      acc0[j] = __builtin_amdgcn_wmma_f32_16x16x32_bf16(
          false, fa0, false, fb, (short)0, acc0[j], false, false);
      acc1[j] = __builtin_amdgcn_wmma_f32_16x16x32_bf16(
          false, fa1, false, fb, (short)0, acc1[j], false, false);
    }
  }

  // Store: f32 C/D layout -> VGPR r holds M = base + r, lanes 16..31 hold +8.
#pragma unroll
  for (int t = 0; t < 2; ++t) {
    const int mBase = tm * 32 + t * 16 + (hi ? 8 : 0);
#pragma unroll
    for (int j = 0; j < NT; ++j) {
      const int n = tng * 16 * NT + lrow + j * 16;
#pragma unroll
      for (int r = 0; r < 8; ++r) {
        const int m = mBase + r;
        float v = (t == 0) ? acc0[j][r] : acc1[j][r];
        if (EPI == EPI_BIAS_SOFTPLUS) v = softplus_f(v + bias[n]);
        else if (EPI == EPI_RESID)    v += resid[(size_t)m * ldres + n];
        C[(size_t)m * ldc + n] = v;
      }
    }
  }
}

// ---------------------------------------------------------------------------
// Embedding gather: x[row, c] = emb[ids[row], c]   (f32)
// ---------------------------------------------------------------------------
__global__ __launch_bounds__(256) void embed_kernel(
    const int* __restrict__ ids, const float* __restrict__ emb,
    float* __restrict__ x, int nrows, int dm) {
  int idx = blockIdx.x * blockDim.x + threadIdx.x;
  if (idx >= nrows * dm) return;
  int r = idx / dm, c = idx - r * dm;
  x[idx] = emb[(size_t)ids[r] * dm + c];
}

// ---------------------------------------------------------------------------
// RMSNorm: one wave32 per row, shfl_xor reduction. Emits bf16 (GEMM A side).
// ---------------------------------------------------------------------------
__global__ __launch_bounds__(256) void rmsnorm_bf16_kernel(
    const float* __restrict__ x, const float* __restrict__ w,
    __bf16* __restrict__ out, int ncols, int nrows) {
  int wave = blockIdx.x * 8 + (threadIdx.x >> 5);
  int lane = threadIdx.x & 31;
  if (wave >= nrows) return;
  const float* xr = x + (size_t)wave * ncols;
  float ss = 0.f;
  for (int c = lane; c < ncols; c += 32) { float v = xr[c]; ss += v * v; }
#pragma unroll
  for (int off = 16; off >= 1; off >>= 1) ss += __shfl_xor(ss, off, 32);
  float scale = rsqrtf(ss / (float)ncols + 1e-5f);
  __bf16* orow = out + (size_t)wave * ncols;
  for (int c = lane; c < ncols; c += 32)
    orow[c] = (__bf16)(xr[c] * scale * w[c]);
}

// ---------------------------------------------------------------------------
// Causal depthwise conv (D_CONV=4) + bias + SiLU. Writes f32 (scan input)
// and bf16 (GEMM A side).
// ---------------------------------------------------------------------------
__global__ __launch_bounds__(256) void conv_silu_kernel(
    const float* __restrict__ xz, const float* __restrict__ cw,
    const float* __restrict__ cb, float* __restrict__ xi,
    __bf16* __restrict__ xi_bf) {
  int idx = blockIdx.x * blockDim.x + threadIdx.x;
  if (idx >= ROWS * D_INNER) return;
  int ch  = idx % D_INNER;
  int row = idx / D_INNER;  // b*L + l
  int l   = row % L_SEQ;
  float acc = cb[ch];
#pragma unroll
  for (int j = 0; j < D_CONV; ++j) {
    int ls = l - (D_CONV - 1) + j;
    if (ls >= 0)
      acc += cw[ch * D_CONV + j] *
             xz[(size_t)(row - (D_CONV - 1) + j) * (2 * D_INNER) + ch];
  }
  float s = silu_f(acc);
  xi[(size_t)row * D_INNER + ch] = s;
  xi_bf[(size_t)row * D_INNER + ch] = (__bf16)s;
}

// ---------------------------------------------------------------------------
// f32 -> bf16 convert (x_dbl copy for the dt GEMM A side).
// ---------------------------------------------------------------------------
__global__ __launch_bounds__(256) void cvt_bf16_kernel(
    const float* __restrict__ in, __bf16* __restrict__ out, int n) {
  int idx = blockIdx.x * blockDim.x + threadIdx.x;
  if (idx >= n) return;
  out[idx] = (__bf16)in[idx];
}

// ---------------------------------------------------------------------------
// Selective scan: one thread per (b, d) channel, 16 states in registers.
//   h_l = exp(max(dt*A, -20)) * h_{l-1} + dt*u*B_l ;  y = C_l . h + u*D
// gated by silu(res); emits bf16 (consumed only by the out-proj GEMM).
// ---------------------------------------------------------------------------
__global__ __launch_bounds__(256) void scan_kernel(
    const float* __restrict__ delta, const float* __restrict__ xi,
    const float* __restrict__ xdbl, const float* __restrict__ xz,
    const float* __restrict__ A_log, const float* __restrict__ Dp,
    __bf16* __restrict__ y) {
  int c = blockIdx.x * blockDim.x + threadIdx.x;
  if (c >= B_SZ * D_INNER) return;
  int b = c / D_INNER, d = c - b * D_INNER;

  float Av[D_STATE], s[D_STATE];
#pragma unroll
  for (int n = 0; n < D_STATE; ++n) {
    Av[n] = -expf(A_log[(size_t)d * D_STATE + n]);
    s[n] = 0.f;
  }
  const float Dv = Dp[d];

  for (int l = 0; l < L_SEQ; ++l) {
    size_t row = (size_t)b * L_SEQ + l;
    float dt = delta[row * D_INNER + d];
    float u  = xi[row * D_INNER + d];
    const float* bc = xdbl + row * XDBL_N;
    float dtu = dt * u;
    float yv = 0.f;
#pragma unroll
    for (int n = 0; n < D_STATE; ++n) {
      float dA = fmaxf(dt * Av[n], -20.f);
      s[n] = expf(dA) * s[n] + dtu * bc[DT_RANK + n];
      yv += bc[DT_RANK + D_STATE + n] * s[n];
    }
    yv += u * Dv;
    float res = xz[row * (2 * D_INNER) + D_INNER + d];
    y[row * D_INNER + d] = (__bf16)(yv * silu_f(res));
  }
}

// ---------------------------------------------------------------------------

static inline int gemm_blocks(int M, int N, int NT) {
  int waves = (M / 32) * (N / (16 * NT));
  return (waves + 7) / 8;
}
static inline int pack_blocks(int KB, int N) {
  int threads = KB * (N / 16) * 32;
  return (threads + 255) / 256;
}

extern "C" void kernel_launch(void* const* d_in, const int* in_sizes, int n_in,
                              void* d_out, int out_size, void* d_ws, size_t ws_size,
                              hipStream_t stream) {
  const int*   ids    = (const int*)  d_in[0];
  const float* emb    = (const float*)d_in[1];
  const float* rms_w  = (const float*)d_in[2];
  const float* in_w   = (const float*)d_in[3];
  const float* cw     = (const float*)d_in[4];
  const float* cb     = (const float*)d_in[5];
  const float* xp_w   = (const float*)d_in[6];
  const float* dt_w   = (const float*)d_in[7];
  const float* dt_b   = (const float*)d_in[8];
  const float* A_log  = (const float*)d_in[9];
  const float* Dp     = (const float*)d_in[10];
  const float* out_w  = (const float*)d_in[11];
  const float* nf_w   = (const float*)d_in[12];
  const float* head_w = (const float*)d_in[13];
  float* out = (float*)d_out;

  // Workspace carve-up, 256B-aligned chunks.
  char* base = (char*)d_ws;
  size_t off = 0;
  auto allocf = [&](size_t n) {
    float* p = (float*)(base + off);
    off = (off + n * sizeof(float) + 255) & ~(size_t)255;
    return p;
  };
  auto allocb = [&](size_t n) {
    __bf16* p = (__bf16*)(base + off);
    off = (off + n * sizeof(__bf16) + 255) & ~(size_t)255;
    return p;
  };
  float*  x0      = allocf((size_t)ROWS * D_MODEL);
  float*  x1      = allocf((size_t)ROWS * D_MODEL);
  float*  xz      = allocf((size_t)ROWS * 2 * D_INNER);
  float*  xi      = allocf((size_t)ROWS * D_INNER);
  float*  xdbl    = allocf((size_t)ROWS * XDBL_N);
  float*  dlt     = allocf((size_t)ROWS * D_INNER);
  __bf16* h_bf    = allocb((size_t)ROWS * D_MODEL);
  __bf16* xi_bf   = allocb((size_t)ROWS * D_INNER);
  __bf16* xdbl_bf = allocb((size_t)ROWS * XDBL_N);
  __bf16* y_bf    = allocb((size_t)ROWS * D_INNER);
  __bf16* wpk     = allocb((size_t)24 * 192 * 512);  // largest: in_w fragments
  (void)ws_size; (void)in_sizes; (void)n_in; (void)out_size;

  float* xcur = x0;
  float* xnext = x1;

  // Embedding
  embed_kernel<<<(ROWS * D_MODEL) / 256, 256, 0, stream>>>(ids, emb, xcur,
                                                           ROWS, D_MODEL);

  for (int i = 0; i < N_LAYER; ++i) {
    const float* lin_w  = in_w  + (size_t)i * D_MODEL * 2 * D_INNER;
    const float* lcw    = cw    + (size_t)i * D_INNER * D_CONV;
    const float* lcb    = cb    + (size_t)i * D_INNER;
    const float* lxp_w  = xp_w  + (size_t)i * D_INNER * XDBL_N;
    const float* ldt_w  = dt_w  + (size_t)i * DT_RANK * D_INNER;
    const float* ldt_b  = dt_b  + (size_t)i * D_INNER;
    const float* lA     = A_log + (size_t)i * D_INNER * D_STATE;
    const float* lD     = Dp    + (size_t)i * D_INNER;
    const float* lout_w = out_w + (size_t)i * D_INNER * D_MODEL;
    const float* lrms   = rms_w + (size_t)i * D_MODEL;

    // h = rmsnorm(x)  (bf16)
    rmsnorm_bf16_kernel<<<ROWS / 8, 256, 0, stream>>>(xcur, lrms, h_bf,
                                                      D_MODEL, ROWS);

    // xz = h @ in_w           (2048 x 3072, K=768, KB=24)
    pack_b_kernel<<<pack_blocks(24, 2 * D_INNER), 256, 0, stream>>>(
        lin_w, wpk, D_MODEL, 2 * D_INNER, 24);
    gemm_wmma_pk<EPI_NONE, 4, 24>
        <<<gemm_blocks(ROWS, 2 * D_INNER, 4), 256, 0, stream>>>(
            h_bf, D_MODEL, wpk, (2 * D_INNER) / 16, xz, 2 * D_INNER,
            ROWS, 2 * D_INNER, nullptr, nullptr, 0);

    // xi = silu(conv(xz[:, :1536]) + cb)   (f32 + bf16)
    conv_silu_kernel<<<(ROWS * D_INNER) / 256, 256, 0, stream>>>(xz, lcw, lcb,
                                                                 xi, xi_bf);

    // x_dbl = xi @ xp_w       (2048 x 80, K=1536, KB=48)
    pack_b_kernel<<<pack_blocks(48, XDBL_N), 256, 0, stream>>>(
        lxp_w, wpk, D_INNER, XDBL_N, 48);
    gemm_wmma_pk<EPI_NONE, 5, 48>
        <<<gemm_blocks(ROWS, XDBL_N, 5), 256, 0, stream>>>(
            xi_bf, D_INNER, wpk, XDBL_N / 16, xdbl, XDBL_N,
            ROWS, XDBL_N, nullptr, nullptr, 0);
    cvt_bf16_kernel<<<(ROWS * XDBL_N) / 256, 256, 0, stream>>>(
        xdbl, xdbl_bf, ROWS * XDBL_N);

    // delta = softplus(x_dbl[:, :48] @ dt_w + dt_b)  (K=48 zero-padded, KB=2)
    pack_b_kernel<<<pack_blocks(2, D_INNER), 256, 0, stream>>>(
        ldt_w, wpk, DT_RANK, D_INNER, 2);
    gemm_wmma_pk<EPI_BIAS_SOFTPLUS, 4, 2>
        <<<gemm_blocks(ROWS, D_INNER, 4), 256, 0, stream>>>(
            xdbl_bf, XDBL_N, wpk, D_INNER / 16, dlt, D_INNER,
            ROWS, D_INNER, ldt_b, nullptr, 0);

    // y = selective_scan(...) * silu(res)   (bf16)
    scan_kernel<<<(B_SZ * D_INNER) / 256, 256, 0, stream>>>(dlt, xi, xdbl, xz,
                                                            lA, lD, y_bf);

    // x_next = y @ out_w + x   (2048 x 768, K=1536, KB=48)
    pack_b_kernel<<<pack_blocks(48, D_MODEL), 256, 0, stream>>>(
        lout_w, wpk, D_INNER, D_MODEL, 48);
    gemm_wmma_pk<EPI_RESID, 4, 48>
        <<<gemm_blocks(ROWS, D_MODEL, 4), 256, 0, stream>>>(
            y_bf, D_INNER, wpk, D_MODEL / 16, xnext, D_MODEL,
            ROWS, D_MODEL, nullptr, xcur, D_MODEL);

    float* t = xcur; xcur = xnext; xnext = t;
  }

  // Final norm + head
  rmsnorm_bf16_kernel<<<ROWS / 8, 256, 0, stream>>>(xcur, nf_w, h_bf, D_MODEL,
                                                    ROWS);
  pack_b_kernel<<<pack_blocks(24, N_MELS), 256, 0, stream>>>(
      head_w, wpk, D_MODEL, N_MELS, 24);
  gemm_wmma_pk<EPI_NONE, 5, 24>
      <<<gemm_blocks(ROWS, N_MELS, 5), 256, 0, stream>>>(
          h_bf, D_MODEL, wpk, N_MELS / 16, out, N_MELS,
          ROWS, N_MELS, nullptr, nullptr, 0);
}
